// SelfAttention_40080634806370
// MI455X (gfx1250) — matrix operand order repaired and checked by reading, not compiled, not run on hardware
//
#include <hip/hip_runtime.h>
#include <hip/hip_bf16.h>

// ---------------------------------------------------------------------------
// Self-attention forward on gfx1250 (MI455X), all matmuls via
// v_wmma_f32_16x16x32_f16 (f16 inputs, f32 accumulate).
//
//   x  : [4, 2048, 512] f32
//   Wq/Wk/Wv : [512, 512] f32   (Linear: y = x @ W.T)
//   out: [4, 2048, 512] f32  laid out [b, t, h, s] flattened
//
// Stage 0: one-shot f32 -> f16 conversion of x and Wq/Wk/Wv into d_ws.
// Stage 1: projections (f16 GEMM) -> Q,K in [b,h,t,s]; V transposed [b,h,s,t].
// Stage 2: flash-attention (online softmax), 16 query rows per wave.
// Workspace: 3*8MB (QKV) + 8MB (x16) + 1.5MB (W16) = 33.5 MB.
// ---------------------------------------------------------------------------

typedef __attribute__((ext_vector_type(16))) _Float16 v16h;
typedef __attribute__((ext_vector_type(8)))  float    v8f;

#define BATCH 4
#define SEQ   2048
#define KDIM  512
#define NHEAD 8
#define HS    64     // head size
#define SCALE 0.044194173824159216f   // 1/sqrt(512)

// A-matrix (16x32 f16) element->K map: lane half selects contiguous K-16 block.
__device__ __forceinline__ int a_kmap(int e, int half) {
    return half * 16 + e;
}

__device__ __forceinline__ v8f wmma_f16(v16h a, v16h b, v8f c) {
    return __builtin_amdgcn_wmma_f32_16x16x32_f16(
        /*neg_a=*/false, a, /*neg_b=*/false, b,
        /*c_mod=*/(short)0, c, /*reuse_a=*/false, /*reuse_b=*/false);
}

// ---------------------------------------------------------------------------
// Stage 0: elementwise f32 -> f16 (vectorized 4-wide)
// ---------------------------------------------------------------------------
__global__ void cvt_f32_to_f16(const float* __restrict__ src,
                               _Float16* __restrict__ dst, int n4) {
    const int i = blockIdx.x * blockDim.x + threadIdx.x;
    if (i < n4) {
        const float4 v = ((const float4*)src)[i];
        _Float16 h0 = (_Float16)v.x, h1 = (_Float16)v.y;
        _Float16 h2 = (_Float16)v.z, h3 = (_Float16)v.w;
        short4 pk;
        pk.x = __builtin_bit_cast(short, h0);
        pk.y = __builtin_bit_cast(short, h1);
        pk.z = __builtin_bit_cast(short, h2);
        pk.w = __builtin_bit_cast(short, h3);
        ((short4*)dst)[i] = pk;
    }
}

// ---------------------------------------------------------------------------
// Stage 1: y = x @ W.T  (one wave computes a 16x64 tile of y), f16 inputs.
// grid: (128 row-supertiles, 8 col-groups of 64, 3 weights), block 128 (4 waves)
// Q,K stored [b,h,t,s]; V stored transposed [b,h,s,t].
// ---------------------------------------------------------------------------
__global__ void proj_f16_wmma(const _Float16* __restrict__ x16,
                              const _Float16* __restrict__ w16,  // 3 x [512,512]
                              _Float16*       __restrict__ qkv) {
    const int lane = threadIdx.x & 31;
    const int wv   = threadIdx.x >> 5;
    const int rt   = blockIdx.x * 4 + wv;     // 0..511 : 16-row tile
    const int cg   = blockIdx.y;              // 0..7   : 64-col group
    const int sel  = blockIdx.z;              // 0..2   : q / k / v

    const _Float16* W   = w16 + (size_t)sel * KDIM * KDIM;
    _Float16*       dst = qkv + (size_t)sel * BATCH * SEQ * KDIM;

    const int M16  = lane & 15;
    const int half = lane >> 4;
    const int row0 = rt * 16;
    const int col0 = cg * 64;

    v8f c0 = {}, c1 = {}, c2 = {}, c3 = {};

    for (int kb = 0; kb < KDIM; kb += 32) {
        // A tile: x16[row0+M, kb .. kb+32)
        v16h a;
#pragma unroll
        for (int e = 0; e < 16; ++e)
            a[e] = x16[(size_t)(row0 + M16) * KDIM + kb + a_kmap(e, half)];

        // Four B tiles: B[k][n] = W[col0 + nt*16 + n][kb + k]
#pragma unroll
        for (int nt = 0; nt < 4; ++nt) {
            v16h b;
#pragma unroll
            for (int e = 0; e < 16; ++e)
                b[e] = W[(size_t)(col0 + nt * 16 + M16) * KDIM +
                         kb + half * 16 + e];
            if (nt == 0) c0 = wmma_f16(a, b, c0);
            else if (nt == 1) c1 = wmma_f16(a, b, c1);
            else if (nt == 2) c2 = wmma_f16(a, b, c2);
            else c3 = wmma_f16(a, b, c3);
        }
    }

#pragma unroll
    for (int nt = 0; nt < 4; ++nt) {
        const v8f c = (nt == 0) ? c0 : (nt == 1) ? c1 : (nt == 2) ? c2 : c3;
#pragma unroll
        for (int r = 0; r < 8; ++r) {
            const int m   = r + 8 * half;
            const int row = row0 + m;                // global row in [0, 8192)
            const int col = col0 + nt * 16 + M16;    // 0..511
            const int b   = row >> 11;               // /2048
            const int t   = row & 2047;
            const int h   = col >> 6;                // /64
            const int s   = col & 63;
            const _Float16 val = (_Float16)c[r];
            if (sel == 2)   // V transposed: [b,h,s,t]
                dst[(((size_t)b * NHEAD + h) * HS + s) * SEQ + t] = val;
            else            // Q,K: [b,h,t,s]
                dst[(((size_t)b * NHEAD + h) * SEQ + t) * HS + s] = val;
        }
    }
}

// ---------------------------------------------------------------------------
// Stage 2: flash attention. One wave = 16 query rows. 32-key blocks.
// grid: (32 b*h, 32 query supertiles of 64), block 128 (4 waves)
// ---------------------------------------------------------------------------
__global__ void attn_flash_wmma(const _Float16* __restrict__ qw,
                                const _Float16* __restrict__ kw,
                                const _Float16* __restrict__ vt,  // [b,h,s,t]
                                float* __restrict__ out) {
    __shared__ _Float16 plds[4][16 * 32];   // per-wave P transpose slab (1 KB)

    const int lane = threadIdx.x & 31;
    const int wv   = threadIdx.x >> 5;
    const int bh   = blockIdx.x;                    // 0..31 = b*8 + h
    const int q0   = (blockIdx.y * 4 + wv) * 16;    // first query row
    const int M16  = lane & 15;
    const int half = lane >> 4;

    const _Float16* qb = qw + (size_t)bh * SEQ * HS;
    const _Float16* kb = kw + (size_t)bh * SEQ * HS;
    const _Float16* vb = vt + (size_t)bh * HS * SEQ;

    // Q A-tiles, k-inner 0..32 and 32..64 (loaded once, reused 64x)
    v16h aq0, aq1;
#pragma unroll
    for (int e = 0; e < 16; ++e) {
        const int kk = a_kmap(e, half);
        aq0[e] = qb[(size_t)(q0 + M16) * HS + kk];
        aq1[e] = qb[(size_t)(q0 + M16) * HS + 32 + kk];
    }

    v8f o0 = {}, o1 = {}, o2 = {}, o3 = {};
    float mrow[8], lrow[8];
#pragma unroll
    for (int r = 0; r < 8; ++r) { mrow[r] = -3.0e38f; lrow[r] = 0.0f; }

    for (int k0 = 0; k0 < SEQ; k0 += 32) {
        if (k0 + 32 < SEQ) {  // uniform branch: EXEC stays all-ones
            __builtin_prefetch(kb + (size_t)(k0 + 32) * HS, 0, 1);
            __builtin_prefetch(vb + (size_t)(k0 + 32), 0, 1);
        }

        // ---- S = Q @ K^T for 32 keys: two 16x16 f32 tiles -----------------
        v8f s0 = {}, s1 = {};
        {
            v16h b00, b01, b10, b11;
#pragma unroll
            for (int e = 0; e < 16; ++e) {
                const int key0 = k0 + M16;        // tile 0 keys
                const int key1 = k0 + 16 + M16;   // tile 1 keys
                const int ki   = half * 16 + e;   // k-inner 0..31
                b00[e] = kb[(size_t)key0 * HS + ki];
                b01[e] = kb[(size_t)key0 * HS + 32 + ki];
                b10[e] = kb[(size_t)key1 * HS + ki];
                b11[e] = kb[(size_t)key1 * HS + 32 + ki];
            }
            s0 = wmma_f16(aq0, b00, s0);
            s0 = wmma_f16(aq1, b01, s0);
            s1 = wmma_f16(aq0, b10, s1);
            s1 = wmma_f16(aq1, b11, s1);
        }

        // ---- online softmax update ---------------------------------------
#pragma unroll
        for (int r = 0; r < 8; ++r) {
            float v0 = s0[r] * SCALE;
            float v1 = s1[r] * SCALE;

            float mx = fmaxf(v0, v1);
#pragma unroll
            for (int off = 8; off >= 1; off >>= 1)     // 16-lane row reduce
                mx = fmaxf(mx, __shfl_xor(mx, off, 32));

            const float mnew  = fmaxf(mrow[r], mx);
            const float alpha = __expf(mrow[r] - mnew);
            mrow[r] = mnew;

            const float p0 = __expf(v0 - mnew);
            const float p1 = __expf(v1 - mnew);
            float rs = p0 + p1;
#pragma unroll
            for (int off = 8; off >= 1; off >>= 1)
                rs += __shfl_xor(rs, off, 32);

            lrow[r] = lrow[r] * alpha + rs;
            o0[r] *= alpha;  o1[r] *= alpha;
            o2[r] *= alpha;  o3[r] *= alpha;

            // stash P (C-layout) into LDS for A-layout reload
            const int m = r + 8 * half;
            plds[wv][m * 32 + M16]      = (_Float16)p0;
            plds[wv][m * 32 + 16 + M16] = (_Float16)p1;
        }

        // ---- reload P in A-layout (16x32 f16) ----------------------------
        v16h pa;
#pragma unroll
        for (int e = 0; e < 16; ++e)
            pa[e] = plds[wv][M16 * 32 + a_kmap(e, half)];

        // ---- O += P @ V: B[k][n] = V^T[s = nt*16+n][key = k0 + k] --------
#pragma unroll
        for (int nt = 0; nt < 4; ++nt) {
            v16h bvv;
#pragma unroll
            for (int e = 0; e < 16; ++e)
                bvv[e] = vb[(size_t)(nt * 16 + M16) * SEQ + k0 + half * 16 + e];
            if (nt == 0) o0 = wmma_f16(pa, bvv, o0);
            else if (nt == 1) o1 = wmma_f16(pa, bvv, o1);
            else if (nt == 2) o2 = wmma_f16(pa, bvv, o2);
            else o3 = wmma_f16(pa, bvv, o3);
        }
    }

    // ---- normalize and write out [b, t, h, s] ----------------------------
    const int b = bh >> 3;
    const int h = bh & 7;
#pragma unroll
    for (int nt = 0; nt < 4; ++nt) {
        const v8f o = (nt == 0) ? o0 : (nt == 1) ? o1 : (nt == 2) ? o2 : o3;
#pragma unroll
        for (int r = 0; r < 8; ++r) {
            const int m = r + 8 * half;
            const int t = q0 + m;
            const int s = nt * 16 + M16;
            out[(((size_t)b * SEQ + t) * NHEAD + h) * HS + s] = o[r] / lrow[r];
        }
    }
}

// ---------------------------------------------------------------------------
extern "C" void kernel_launch(void* const* d_in, const int* in_sizes, int n_in,
                              void* d_out, int out_size, void* d_ws, size_t ws_size,
                              hipStream_t stream) {
    const float* x  = (const float*)d_in[0];
    const float* Wq = (const float*)d_in[1];
    const float* Wk = (const float*)d_in[2];
    const float* Wv = (const float*)d_in[3];
    float* out = (float*)d_out;

    const size_t perMat = (size_t)BATCH * SEQ * KDIM;   // 4,194,304 elems
    const size_t wElems = (size_t)KDIM * KDIM;          //   262,144 elems

    _Float16* qkv = (_Float16*)d_ws;              // 3 * perMat
    _Float16* x16 = qkv + 3 * perMat;             // perMat
    _Float16* w16 = x16 + perMat;                 // 3 * wElems

    // Stage 0: one-shot f32 -> f16 conversions
    {
        const int n4x = (int)(perMat / 4);
        cvt_f32_to_f16<<<(n4x + 255) / 256, 256, 0, stream>>>(x, x16, n4x);
        const int n4w = (int)(wElems / 4);
        cvt_f32_to_f16<<<(n4w + 255) / 256, 256, 0, stream>>>(Wq, w16, n4w);
        cvt_f32_to_f16<<<(n4w + 255) / 256, 256, 0, stream>>>(Wk, w16 + wElems, n4w);
        cvt_f32_to_f16<<<(n4w + 255) / 256, 256, 0, stream>>>(Wv, w16 + 2 * wElems, n4w);
    }

    // Stage 1: Q,K,V projections (f16 GEMM); V stored transposed
    dim3 g1(128, 8, 3);
    proj_f16_wmma<<<g1, 128, 0, stream>>>(x16, w16, qkv);

    // Stage 2: flash attention
    const _Float16* qw = qkv;
    const _Float16* kw = qkv + perMat;
    const _Float16* vt = qkv + 2 * perMat;
    dim3 g2(BATCH * NHEAD, SEQ / 64, 1);
    attn_flash_wmma<<<g2, 128, 0, stream>>>(qw, kw, vt, out);
}
